// MambaBlock_82849919140302
// MI455X (gfx1250) — matrix operand order repaired and checked by reading, not compiled, not run on hardware
//
#include <hip/hip_runtime.h>
#include <hip/hip_bf16.h>
#include <math.h>

#define CHN       128
#define D_STATE   64
#define D_CONVW   5
#define D_INNER   256
#define NHEADS    4
#define HEADDIM   64
#define CONV_DIM  384
#define D_IN_PROJ 644
#define B_SZ      4
#define HWPIX     4096
#define NPIX      (B_SZ * HWPIX)   // 16384
#define BROW      72               // padded LDS row stride (halves) for B tile

typedef __attribute__((ext_vector_type(16))) _Float16 v16h;
typedef __attribute__((ext_vector_type(8)))  _Float16 v8h;
typedef __attribute__((ext_vector_type(8)))  float    v8f;

__device__ __forceinline__ float silu_f(float x) { return x / (1.f + __expf(-x)); }

// ---------------- f32 -> f16 weight conversion ----------------
__global__ void cvt_f16_kernel(const float* __restrict__ src, _Float16* __restrict__ dst, int n) {
    int i = blockIdx.x * blockDim.x + threadIdx.x;
    if (i < n) dst[i] = (_Float16)src[i];
}

// ---------------- LayerNorm over C=128 per pixel, emit f16 A matrix ----------------
__global__ __launch_bounds__(128) void ln_kernel(const float* __restrict__ noisy,
                                                 const float* __restrict__ lnw,
                                                 const float* __restrict__ lnb,
                                                 _Float16* __restrict__ Aout) {
    int pix = blockIdx.x;            // b*4096 + yx
    int b   = pix >> 12;
    int yx  = pix & 4095;
    int c   = threadIdx.x;
    float v = noisy[((size_t)(b * CHN + c) << 12) + yx];
    float sum = v, sq = v * v;
    for (int o = 16; o > 0; o >>= 1) {
        sum += __shfl_down(sum, o, 32);
        sq  += __shfl_down(sq,  o, 32);
    }
    __shared__ float s1[4], s2[4];
    int w = threadIdx.x >> 5;
    if ((threadIdx.x & 31) == 0) { s1[w] = sum; s2[w] = sq; }
    __syncthreads();
    sum = s1[0] + s1[1] + s1[2] + s1[3];
    sq  = s2[0] + s2[1] + s2[2] + s2[3];
    float mu  = sum * (1.f / 128.f);
    float var = sq * (1.f / 128.f) - mu * mu;
    float nv  = (v - mu) * rsqrtf(var + 1e-5f) * lnw[c] + lnb[c];
    Aout[(size_t)pix * CHN + c] = (_Float16)nv;
}

// ---- async copy of one 64(N) x 64(K) f16 B tile into LDS (padded rows) ----
// 512 chunks of 16B; each of 256 threads issues 2 async b128 copies.
__device__ __forceinline__ void async_copy_b_tile(const _Float16* __restrict__ B,
                                                  int N, int K, int n0, int k0,
                                                  _Float16* sbase, int t) {
#pragma unroll
    for (int c = t; c < 512; c += 256) {
        int n  = c >> 3;
        int ko = (c & 7) * 8;                 // half offset within row
        int gn = n0 + n; if (gn >= N) gn = N - 1;   // clamp reads; stores guarded later
        const _Float16* gp = B + (size_t)gn * K + k0 + ko;
        unsigned lds = (unsigned)(size_t)(sbase + n * BROW + ko);
        asm volatile("global_load_async_to_lds_b128 %0, %1, off"
                     :: "v"(lds), "v"(gp) : "memory");
    }
}

// ---------------- WMMA GEMM: C[M,N] = A[M,K] * B[N,K]^T (+bias, relu) ----------
// block = 256 threads (8 waves); block tile = 128 M x 64 N; K staged 64-wide,
// double-buffered in LDS via async global->LDS copies (ASYNCcnt).
// Requires: M % 128 == 0, K % 64 == 0.
__global__ __launch_bounds__(256) void wmma_gemm_kernel(const _Float16* __restrict__ A,
                                                        const _Float16* __restrict__ B,
                                                        float* __restrict__ C,
                                                        int M, int N, int K,
                                                        const float* __restrict__ bias,
                                                        int relu) {
    __shared__ _Float16 sB[2][64 * BROW];
    const int t     = threadIdx.x;
    const int lane  = t & 31;
    const int wave  = t >> 5;
    const int m0    = blockIdx.y * 128 + wave * 16;
    const int n0    = blockIdx.x * 64;
    const int l16   = lane & 15;
    const int khalf = lane >> 4;          // 0 or 1

    v8f acc0 = {}, acc1 = {}, acc2 = {}, acc3 = {};
    const _Float16* arow = A + (size_t)(m0 + l16) * K + khalf * 8;

    async_copy_b_tile(B, N, K, n0, 0, sB[0], t);

    for (int kt = 0; kt < K; kt += 64) {
        const int cur = (kt >> 6) & 1;
        if (kt + 64 < K) {
            async_copy_b_tile(B, N, K, n0, kt + 64, sB[cur ^ 1], t);
            asm volatile("s_wait_asynccnt 0x2" ::: "memory");   // wait current stage only
        } else {
            asm volatile("s_wait_asynccnt 0x0" ::: "memory");
        }
        __syncthreads();

        const _Float16* sb = sB[cur];
#pragma unroll
        for (int kk = 0; kk < 64; kk += 32) {
            union { v16h v; v8h h[2]; } af;
            af.h[0] = *(const v8h*)(arow + kt + kk);
            af.h[1] = *(const v8h*)(arow + kt + kk + 16);
            __builtin_prefetch(arow + kt + kk + 64, 0, 1);
#pragma unroll
            for (int nt = 0; nt < 4; ++nt) {
                const _Float16* bp = sb + (nt * 16 + l16) * BROW + kk + khalf * 8;
                union { v16h v; v8h h[2]; } bf;
                bf.h[0] = *(const v8h*)(bp);
                bf.h[1] = *(const v8h*)(bp + 16);
                if (nt == 0)
                    acc0 = __builtin_amdgcn_wmma_f32_16x16x32_f16(false, af.v, false, bf.v, (short)0, acc0, false, false);
                else if (nt == 1)
                    acc1 = __builtin_amdgcn_wmma_f32_16x16x32_f16(false, af.v, false, bf.v, (short)0, acc1, false, false);
                else if (nt == 2)
                    acc2 = __builtin_amdgcn_wmma_f32_16x16x32_f16(false, af.v, false, bf.v, (short)0, acc2, false, false);
                else
                    acc3 = __builtin_amdgcn_wmma_f32_16x16x32_f16(false, af.v, false, bf.v, (short)0, acc3, false, false);
            }
        }
        __syncthreads();
    }

    v8f accs[4] = {acc0, acc1, acc2, acc3};
#pragma unroll
    for (int nt = 0; nt < 4; ++nt) {
        int n = n0 + nt * 16 + l16;
        if (n >= N) continue;
        float bv = bias ? bias[n] : 0.f;
#pragma unroll
        for (int v = 0; v < 8; ++v) {
            int m = m0 + v + 8 * khalf;
            float val = accs[nt][v] + bv;
            if (relu) val = fmaxf(val, 0.f);
            C[(size_t)m * N + n] = val;
        }
    }
}

// ---------------- dt = softplus(raw + dt_bias) ----------------
__global__ void dt_kernel(const float* __restrict__ zxbcdt,
                          const float* __restrict__ dt_bias,
                          float* __restrict__ dtbuf, int n) {
    int i = blockIdx.x * blockDim.x + threadIdx.x;  // i = pix*4 + h
    if (i >= n) return;
    int pix = i >> 2, h = i & 3;
    float x = zxbcdt[(size_t)pix * D_IN_PROJ + 640 + h] + dt_bias[h];
    dtbuf[i] = (x > 20.f) ? x : log1pf(__expf(x));
}

// ---------------- causal depthwise conv1d + silu on xBC ----------------
__global__ void conv1d_kernel(const float* __restrict__ zxbcdt,
                              const float* __restrict__ conv_w,
                              const float* __restrict__ conv_b,
                              float* __restrict__ xBCc, int n) {
    int i = blockIdx.x * blockDim.x + threadIdx.x;  // i = pix*384 + ch
    if (i >= n) return;
    int pix = i / CONV_DIM, ch = i - pix * CONV_DIM;
    int b = pix >> 12, l = pix & 4095;
    float acc = conv_b[ch];
#pragma unroll
    for (int k = 0; k < D_CONVW; ++k) {
        int src = l + k - (D_CONVW - 1);
        if (src >= 0)
            acc += zxbcdt[((size_t)(b * HWPIX + src)) * D_IN_PROJ + D_INNER + ch] * conv_w[ch * D_CONVW + k];
    }
    xBCc[(size_t)pix * CONV_DIM + ch] = silu_f(acc);
}

// ---------------- sequential SSM scan: one block per (b, head) ----------------
__global__ __launch_bounds__(256) void scan_kernel(const float* __restrict__ xBCc,
                                                   const float* __restrict__ dtbuf,
                                                   const float* __restrict__ A_log,
                                                   const float* __restrict__ Dp,
                                                   float* __restrict__ ybuf) {
    const int b = blockIdx.x / NHEADS;
    const int h = blockIdx.x % NHEADS;
    const int t = threadIdx.x;
    const int p  = t >> 2;            // 0..63 : headdim index
    const int nq = t & 3;             // quarter of d_state
    const int n0 = nq * 16;
    const float negAexp = -__expf(A_log[h]);
    const float Dh = Dp[h];

    float hreg[16];
#pragma unroll
    for (int j = 0; j < 16; ++j) hreg[j] = 0.f;

    __shared__ float sx[64], sB[64], sC[64], sdt[2];

    for (int l = 0; l < HWPIX; ++l) {
        const size_t base = ((size_t)(b * HWPIX + l)) * CONV_DIM;
        if (t < 64)        sx[t]       = xBCc[base + h * HEADDIM + t];
        else if (t < 128)  sB[t - 64]  = xBCc[base + D_INNER + (t - 64)];
        else if (t < 192)  sC[t - 128] = xBCc[base + D_INNER + D_STATE + (t - 128)];
        else if (t == 192) {
            float dt = dtbuf[(size_t)(b * HWPIX + l) * NHEADS + h];
            sdt[0] = dt;
            sdt[1] = __expf(negAexp * dt);
        }
        __syncthreads();
        const float dt = sdt[0], dA = sdt[1];
        const float xp = sx[p];
        const float coef = dt * xp;
        float acc = 0.f;
#pragma unroll
        for (int j = 0; j < 16; ++j) {
            hreg[j] = hreg[j] * dA + coef * sB[n0 + j];
            acc += hreg[j] * sC[n0 + j];
        }
        acc += __shfl_xor(acc, 1, 32);
        acc += __shfl_xor(acc, 2, 32);
        if (nq == 0)
            ybuf[(size_t)(b * HWPIX + l) * D_INNER + h * HEADDIM + p] = acc + Dh * xp;
        __syncthreads();
    }
}

// ---------------- y *= silu(z); RMSNorm(256); emit f16 A2 ----------------
__global__ __launch_bounds__(256) void gate_rms_kernel(const float* __restrict__ ybuf,
                                                       const float* __restrict__ zxbcdt,
                                                       const float* __restrict__ rms_w,
                                                       _Float16* __restrict__ A2) {
    int pix = blockIdx.x;
    int i   = threadIdx.x;
    float z = zxbcdt[(size_t)pix * D_IN_PROJ + i];
    float g = ybuf[(size_t)pix * D_INNER + i] * silu_f(z);
    float sq = g * g;
    for (int o = 16; o > 0; o >>= 1) sq += __shfl_down(sq, o, 32);
    __shared__ float s[8];
    int w = i >> 5;
    if ((i & 31) == 0) s[w] = sq;
    __syncthreads();
    float tot = 0.f;
#pragma unroll
    for (int k = 0; k < 8; ++k) tot += s[k];
    float scale = rsqrtf(tot * (1.f / 256.f) + 1e-5f);
    A2[(size_t)pix * D_INNER + i] = (_Float16)(g * scale * rms_w[i]);
}

// ---------------- res[NCHW] = noisy + m ----------------
__global__ void residual_kernel(const float* __restrict__ noisy,
                                const float* __restrict__ mbuf,
                                float* __restrict__ res, int n) {
    int idx = blockIdx.x * blockDim.x + threadIdx.x;
    if (idx >= n) return;
    int b  = idx >> 19;
    int c  = (idx >> 12) & 127;
    int yx = idx & 4095;
    int pix = (b << 12) | yx;
    res[idx] = noisy[idx] + mbuf[(size_t)pix * CHN + c];
}

__device__ __forceinline__ int reflect_idx(int v) {
    if (v < 0) return -v;
    if (v > 63) return 126 - v;
    return v;
}

// ---------------- reflect-pad im2col to f16 [pix][ci*9 + ky*3 + kx] ----------------
// mode 0: src is NCHW f32 (res).  mode 1: src is [pix][128] f32 (ff1).
__global__ void im2col_kernel(const float* __restrict__ src,
                              _Float16* __restrict__ dst, int mode, int n) {
    int i = blockIdx.x * blockDim.x + threadIdx.x;  // i = pix*128 + ci
    if (i >= n) return;
    int pix = i >> 7, ci = i & 127;
    int b = pix >> 12, yx = pix & 4095;
    int y = yx >> 6, x = yx & 63;
    _Float16* o = dst + (size_t)pix * 1152 + ci * 9;
#pragma unroll
    for (int ky = 0; ky < 3; ++ky) {
        int ry = reflect_idx(y + ky - 1);
#pragma unroll
        for (int kx = 0; kx < 3; ++kx) {
            int rx = reflect_idx(x + kx - 1);
            float v;
            if (mode == 0)
                v = src[((size_t)(b * CHN + ci) << 12) + (ry << 6) + rx];
            else
                v = src[((size_t)((b << 12) + (ry << 6) + rx)) * CHN + ci];
            o[ky * 3 + kx] = (_Float16)v;
        }
    }
}

// ---------------- out = res + ff2 ; append aux ----------------
__global__ void final_kernel(const float* __restrict__ res,
                             const float* __restrict__ ff2,
                             const float* __restrict__ aux,
                             float* __restrict__ out, int n) {
    int idx = blockIdx.x * blockDim.x + threadIdx.x;
    if (idx >= n) return;
    int b  = idx >> 19;
    int c  = (idx >> 12) & 127;
    int yx = idx & 4095;
    int pix = (b << 12) | yx;
    out[idx] = res[idx] + ff2[(size_t)pix * CHN + c];
    if (idx == 0) out[n] = aux[0];
}

extern "C" void kernel_launch(void* const* d_in, const int* in_sizes, int n_in,
                              void* d_out, int out_size, void* d_ws, size_t ws_size,
                              hipStream_t stream) {
    const float* noisy      = (const float*)d_in[0];
    const float* aux        = (const float*)d_in[1];
    const float* ln_w       = (const float*)d_in[2];
    const float* ln_b       = (const float*)d_in[3];
    const float* in_proj_w  = (const float*)d_in[4];
    const float* conv_w     = (const float*)d_in[5];
    const float* conv_b     = (const float*)d_in[6];
    const float* dt_bias    = (const float*)d_in[7];
    const float* A_log      = (const float*)d_in[8];
    const float* Dvec       = (const float*)d_in[9];
    const float* rms_w      = (const float*)d_in[10];
    const float* out_proj_w = (const float*)d_in[11];
    const float* ff_w1      = (const float*)d_in[12];
    const float* ff_b1      = (const float*)d_in[13];
    const float* ff_w2      = (const float*)d_in[14];
    const float* ff_b2      = (const float*)d_in[15];
    float* out = (float*)d_out;

    char* ws = (char*)d_ws;
    size_t off = 0;
    auto alloc = [&](size_t bytes) -> char* {
        char* p = ws + off;
        off = (off + bytes + 255) & ~(size_t)255;
        return p;
    };

    _Float16* WipH   = (_Float16*)alloc((size_t)D_IN_PROJ * CHN * 2);
    _Float16* WopH   = (_Float16*)alloc((size_t)CHN * D_INNER * 2);
    _Float16* Wf1H   = (_Float16*)alloc((size_t)CHN * CHN * 9 * 2);
    _Float16* Wf2H   = (_Float16*)alloc((size_t)CHN * CHN * 9 * 2);
    _Float16* Anorm  = (_Float16*)alloc((size_t)NPIX * CHN * 2);
    float*    zxbcdt = (float*)   alloc((size_t)NPIX * D_IN_PROJ * 4);
    float*    xBCc   = (float*)   alloc((size_t)NPIX * CONV_DIM * 4);
    float*    dtbuf  = (float*)   alloc((size_t)NPIX * NHEADS * 4);
    float*    ybuf   = (float*)   alloc((size_t)NPIX * D_INNER * 4);
    _Float16* A2     = (_Float16*)alloc((size_t)NPIX * D_INNER * 2);
    float*    mbuf   = (float*)   alloc((size_t)NPIX * CHN * 4);
    float*    res    = (float*)   alloc((size_t)NPIX * CHN * 4);
    _Float16* A3     = (_Float16*)alloc((size_t)NPIX * 1152 * 2);
    float*    ff1    = (float*)   alloc((size_t)NPIX * CHN * 4);
    float*    ff2    = (float*)   alloc((size_t)NPIX * CHN * 4);
    (void)ws_size; (void)n_in; (void)in_sizes; (void)out_size;

    // weight conversions to f16
    cvt_f16_kernel<<<(D_IN_PROJ * CHN + 255) / 256, 256, 0, stream>>>(in_proj_w, WipH, D_IN_PROJ * CHN);
    cvt_f16_kernel<<<(CHN * D_INNER + 255) / 256, 256, 0, stream>>>(out_proj_w, WopH, CHN * D_INNER);
    cvt_f16_kernel<<<(CHN * CHN * 9 + 255) / 256, 256, 0, stream>>>(ff_w1, Wf1H, CHN * CHN * 9);
    cvt_f16_kernel<<<(CHN * CHN * 9 + 255) / 256, 256, 0, stream>>>(ff_w2, Wf2H, CHN * CHN * 9);

    // LayerNorm -> f16
    ln_kernel<<<NPIX, 128, 0, stream>>>(noisy, ln_w, ln_b, Anorm);

    // in_proj GEMM: [16384,644] = A[16384,128] x W[644,128]^T
    {
        dim3 g((D_IN_PROJ + 63) / 64, NPIX / 128);
        wmma_gemm_kernel<<<g, 256, 0, stream>>>(Anorm, WipH, zxbcdt, NPIX, D_IN_PROJ, CHN, nullptr, 0);
    }

    // dt softplus
    dt_kernel<<<(NPIX * NHEADS + 255) / 256, 256, 0, stream>>>(zxbcdt, dt_bias, dtbuf, NPIX * NHEADS);

    // depthwise causal conv + silu
    conv1d_kernel<<<(NPIX * CONV_DIM + 255) / 256, 256, 0, stream>>>(zxbcdt, conv_w, conv_b, xBCc, NPIX * CONV_DIM);

    // sequential scan: 16 blocks = (batch x heads)
    scan_kernel<<<B_SZ * NHEADS, 256, 0, stream>>>(xBCc, dtbuf, A_log, Dvec, ybuf);

    // gate + RMSNorm -> f16
    gate_rms_kernel<<<NPIX, 256, 0, stream>>>(ybuf, zxbcdt, rms_w, A2);

    // out_proj GEMM: [16384,128] = A2[16384,256] x W[128,256]^T
    {
        dim3 g((CHN + 63) / 64, NPIX / 128);
        wmma_gemm_kernel<<<g, 256, 0, stream>>>(A2, WopH, mbuf, NPIX, CHN, D_INNER, nullptr, 0);
    }

    // residual
    residual_kernel<<<(NPIX * CHN + 255) / 256, 256, 0, stream>>>(noisy, mbuf, res, NPIX * CHN);

    // conv1 (implicit GEMM): im2col(res) -> A3 ; ff1 = relu(A3 x Wf1^T + b1)
    im2col_kernel<<<(NPIX * CHN + 255) / 256, 256, 0, stream>>>(res, A3, 0, NPIX * CHN);
    {
        dim3 g((CHN + 63) / 64, NPIX / 128);
        wmma_gemm_kernel<<<g, 256, 0, stream>>>(A3, Wf1H, ff1, NPIX, CHN, 1152, ff_b1, 1);
    }

    // conv2: im2col(ff1) -> A3 (reused) ; ff2 = relu(A3 x Wf2^T + b2)
    im2col_kernel<<<(NPIX * CHN + 255) / 256, 256, 0, stream>>>(ff1, A3, 1, NPIX * CHN);
    {
        dim3 g((CHN + 63) / 64, NPIX / 128);
        wmma_gemm_kernel<<<g, 256, 0, stream>>>(A3, Wf2H, ff2, NPIX, CHN, 1152, ff_b2, 1);
    }

    // out = res + ff2 ; aux passthrough
    final_kernel<<<(NPIX * CHN + 255) / 256, 256, 0, stream>>>(res, ff2, aux, out, NPIX * CHN);
}
